// QuasiRNNLayer_27771258536091
// MI455X (gfx1250) — matrix operand order repaired and checked by reading, not compiled, not run on hardware
//
#include <hip/hip_runtime.h>
#include <hip/hip_bf16.h>

// ---------------------------------------------------------------------------
// QuasiRNN layer for MI455X (gfx1250, wave32, WMMA).
//   B=32, T=1024, D=512, H=512, N3=3H=1536, M=B*T=32768
// Phase 1: f32->bf16 convert x,W (gates feed saturating nonlinearities, so
//          bf16 GEMM inputs are precision-safe; halves streamed bytes).
// Phase 2: y = x @ W^T + b via v_wmma_f32_16x16x32_bf16; branchless fused
//          activations (tanh/sigmoid unified as A*rcp(1+exp(B*v))+C).
// Phase 3: segmented affine scan (8 segments x 128 steps) for 8x parallelism:
//          pass1 per-segment (p, P=prod(1-f)), combine chains segment inits,
//          pass2 re-runs segments with true inits and streams final outputs.
// ---------------------------------------------------------------------------

typedef __attribute__((ext_vector_type(16))) __bf16 v16bf;
typedef __attribute__((ext_vector_type(8)))  __bf16 v8bf;
typedef __attribute__((ext_vector_type(8)))  float  v8f;
typedef __attribute__((ext_vector_type(4)))  float  v4f;

#define QB    32
#define QT    1024
#define QD    512
#define QH    512
#define QN3   1536
#define QM    32768            // B*T
#define NSEG  8
#define SEGLEN (QT / NSEG)     // 128
#define NCHAIN (2 * QB * QH)   // 32768 chains: dir x batch x hidden

// ---- f32 -> bf16 conversion, 8 elements / thread -------------------------
__global__ __launch_bounds__(256) void qrnn_cvt_bf16(const float* __restrict__ in,
                                                     __bf16* __restrict__ out,
                                                     int n8) {
    int i = blockIdx.x * blockDim.x + threadIdx.x;
    if (i >= n8) return;
    const v4f* p = (const v4f*)in + (size_t)i * 2;
    v4f a = __builtin_nontemporal_load(p);
    v4f b = __builtin_nontemporal_load(p + 1);
    v8bf o;
    o[0] = (__bf16)a[0]; o[1] = (__bf16)a[1]; o[2] = (__bf16)a[2]; o[3] = (__bf16)a[3];
    o[4] = (__bf16)b[0]; o[5] = (__bf16)b[1]; o[6] = (__bf16)b[2]; o[7] = (__bf16)b[3];
    *(v8bf*)(out + (size_t)i * 8) = o;
}

// ---- WMMA GEMM + fused bias/activation -----------------------------------
// One wave per 64x64 tile of Y[M][N3]; all 8 waves of a block share m0 so the
// A tile is served from WGP$ once per block.  4x4 grid of 16x16x32 bf16 WMMAs.
__global__ __launch_bounds__(256) void qrnn_gemm_gates(
        const __bf16* __restrict__ A,     // [QM][QD]
        const __bf16* __restrict__ Wb,    // [QN3][QD]
        const float*  __restrict__ bias,  // [QN3]
        float* __restrict__ Y)            // [QM][QN3] activated gates
{
    const int lane = threadIdx.x & 31;
    const int wave = threadIdx.x >> 5;
    const int mgrp = blockIdx.x / 3;                 // 512 m-tiles
    const int ngrp = blockIdx.x % 3;                 // 3 groups of 8 n-tiles
    const int m0 = mgrp * 64;
    const int n0 = (ngrp * 8 + wave) * 64;

    const int ll = lane & 15;       // row/col within 16
    const int lh = lane >> 4;       // half-wave select

    v8f acc[4][4];
#pragma unroll
    for (int i = 0; i < 4; ++i)
#pragma unroll
        for (int j = 0; j < 4; ++j)
            acc[i][j] = (v8f)0.0f;

    for (int k0 = 0; k0 < QD; k0 += 32) {
        // A fragment (16x32 bf16, ISA 7.12.2 wave32 layout): lane holds row
        // ll, K chunks {kb..kb+7} and {kb+16..kb+23}, kb = k0 + lh*8
        v16bf af[4];
#pragma unroll
        for (int i = 0; i < 4; ++i) {
            const __bf16* pa = A + (size_t)(m0 + i * 16 + ll) * QD + k0 + lh * 8;
            ((v8bf*)&af[i])[0] = *(const v8bf*)(pa);
            ((v8bf*)&af[i])[1] = *(const v8bf*)(pa + 16);
        }
        // B fragment (32x16 bf16): lane holds col ll, K chunks {kb..kb+7} and {kb+16..kb+23}, kb = k0 + lh*8
        v16bf bf[4];
#pragma unroll
        for (int j = 0; j < 4; ++j) {
            const __bf16* pb = Wb + (size_t)(n0 + j * 16 + ll) * QD + k0 + lh * 8;
            ((v8bf*)&bf[j])[0] = *(const v8bf*)(pb);
            ((v8bf*)&bf[j])[1] = *(const v8bf*)(pb + 16);
        }
#pragma unroll
        for (int i = 0; i < 4; ++i)
#pragma unroll
            for (int j = 0; j < 4; ++j)
                acc[i][j] = __builtin_amdgcn_wmma_f32_16x16x32_bf16(
                    false, af[i], false, bf[j], (short)0, acc[i][j], false, false);
    }

    // Branchless activation: act = Ac*rcp(1 + exp(Bc*v)) + Cc  (wave-uniform)
    //   tanh(v)    = 1 - 2*rcp(1+exp(2v))   -> Ac=-2, Bc= 2, Cc=1
    //   sigmoid(v) =     rcp(1+exp(-v))     -> Ac= 1, Bc=-1, Cc=0
    const bool is_z = (n0 < QH);
    const float Ac = is_z ? -2.0f : 1.0f;
    const float Bc = is_z ?  2.0f : -1.0f;
    const float Cc = is_z ?  1.0f : 0.0f;

    float bv[4];
#pragma unroll
    for (int j = 0; j < 4; ++j) bv[j] = bias[n0 + j * 16 + ll];

    // C/D layout: lane l, vgpr r -> (M = r + lh*8, N = ll) per 16x16 subtile
#pragma unroll
    for (int i = 0; i < 4; ++i) {
#pragma unroll
        for (int j = 0; j < 4; ++j) {
#pragma unroll
            for (int r = 0; r < 8; ++r) {
                float v = acc[i][j][r] + bv[j];
                float e = __expf(Bc * v);
                float act = fmaf(Ac, __builtin_amdgcn_rcpf(1.0f + e), Cc);
                int row = m0 + i * 16 + r + lh * 8;
                int col = n0 + j * 16 + ll;
                Y[(size_t)row * QN3 + col] = act;
            }
        }
    }
}

// ---- pass 1: per-segment zero-init scan result p and decay product P ------
// thread -> (seg, dir, b, j); 262144 threads, 128 steps each.
__global__ __launch_bounds__(256) void qrnn_seg_scan(
        const float* __restrict__ Y,      // [QM][QN3] activated gates
        float* __restrict__ p_end,        // [NSEG][NCHAIN]
        float* __restrict__ P_end)        // [NSEG][NCHAIN]
{
    int tid  = blockIdx.x * blockDim.x + threadIdx.x;
    int j    = tid & 511;
    int b    = (tid >> 9) & 31;
    int dir  = (tid >> 14) & 1;
    int seg  = tid >> 15;
    int chain = tid & (NCHAIN - 1);
    const float* base = Y + (size_t)b * QT * QN3 + j;

    float p = 0.0f, P = 1.0f;
#pragma unroll 4
    for (int i = 0; i < SEGLEN; ++i) {
        int u = seg * SEGLEN + i;
        int t = dir ? (QT - 1 - u) : u;
        const float* g = base + (size_t)t * QN3;
        if (i + 16 < SEGLEN) {
            const float* q = base + (size_t)(dir ? t - 16 : t + 16) * QN3;
            __builtin_prefetch(q, 0, 3);
            __builtin_prefetch(q + QH, 0, 3);
        }
        float z = g[0];
        float f = g[QH];
        float om = 1.0f - f;
        p = f * z + om * p;
        P = P * om;
    }
    p_end[(size_t)seg * NCHAIN + chain] = p;
    P_end[(size_t)seg * NCHAIN + chain] = P;
}

// ---- combine: chain segment summaries into per-segment initial states -----
__global__ __launch_bounds__(256) void qrnn_seg_combine(
        const float* __restrict__ p_end,
        const float* __restrict__ P_end,
        float* __restrict__ seg_init)     // [NSEG][NCHAIN]
{
    int c = blockIdx.x * blockDim.x + threadIdx.x;   // chain id
    float H = 0.0f;
#pragma unroll
    for (int s = 0; s < NSEG; ++s) {
        seg_init[(size_t)s * NCHAIN + c] = H;
        H = p_end[(size_t)s * NCHAIN + c] + P_end[(size_t)s * NCHAIN + c] * H;
    }
}

// ---- pass 2: re-run segments with true initial state, stream outputs ------
__global__ __launch_bounds__(256) void qrnn_seg_emit(
        const float* __restrict__ Y,        // [QM][QN3]
        const float* __restrict__ seg_init, // [NSEG][NCHAIN]
        float* __restrict__ h_out,          // [QB][QT][2*QH]
        float* __restrict__ c_last)         // [QT][2*QH]
{
    int tid  = blockIdx.x * blockDim.x + threadIdx.x;
    int j    = tid & 511;
    int b    = (tid >> 9) & 31;
    int dir  = (tid >> 14) & 1;
    int seg  = tid >> 15;
    int chain = tid & (NCHAIN - 1);
    const float* base = Y + (size_t)b * QT * QN3 + j;
    const int colofs = dir * QH;

    float state = seg_init[(size_t)seg * NCHAIN + chain];
#pragma unroll 4
    for (int i = 0; i < SEGLEN; ++i) {
        int u = seg * SEGLEN + i;
        int t = dir ? (QT - 1 - u) : u;
        const float* g = base + (size_t)t * QN3;
        if (i + 16 < SEGLEN) {
            const float* q = base + (size_t)(dir ? t - 16 : t + 16) * QN3;
            __builtin_prefetch(q, 0, 3);
            __builtin_prefetch(q + QH, 0, 3);
            __builtin_prefetch(q + 2 * QH, 0, 3);
        }
        float z = g[0];
        float f = g[QH];
        float o = g[2 * QH];
        state = f * z + (1.0f - f) * state;
        // streamed once, never re-read -> non-temporal stores keep L2 for gates
        __builtin_nontemporal_store(o * state,
            h_out + ((size_t)b * QT + t) * (2 * QH) + colofs + j);
        if (b == QB - 1)
            __builtin_nontemporal_store(state,
                c_last + (size_t)t * (2 * QH) + colofs + j);
    }
}

extern "C" void kernel_launch(void* const* d_in, const int* in_sizes, int n_in,
                              void* d_out, int out_size, void* d_ws, size_t ws_size,
                              hipStream_t stream) {
    const float* x    = (const float*)d_in[0];  // [32,1024,512]
    const float* W    = (const float*)d_in[1];  // [1536,512]
    const float* bias = (const float*)d_in[2];  // [1536]

    // workspace layout (~229 MB)
    char* ws = (char*)d_ws;
    size_t off = 0;
    __bf16* xb = (__bf16*)(ws + off); off += (size_t)QM * QD * 2;      // 32 MB
    __bf16* wb = (__bf16*)(ws + off); off += (size_t)QN3 * QD * 2;     // 1.5 MB
    float*  yb = (float*)(ws + off);  off += (size_t)QM * QN3 * 4;     // 192 MB
    float*  p_end    = (float*)(ws + off); off += (size_t)NSEG * NCHAIN * 4;
    float*  P_end    = (float*)(ws + off); off += (size_t)NSEG * NCHAIN * 4;
    float*  seg_init = (float*)(ws + off); off += (size_t)NSEG * NCHAIN * 4;

    float* h_out  = (float*)d_out;                               // [32][1024][1024]
    float* c_last = (float*)d_out + (size_t)QB * QT * 2 * QH;    // [1024][1024]

    // 1) convert x, W to bf16
    {
        int n8 = (QM * QD) / 8;
        qrnn_cvt_bf16<<<n8 / 256, 256, 0, stream>>>(x, xb, n8);
    }
    {
        int n8 = (QN3 * QD) / 8;
        qrnn_cvt_bf16<<<n8 / 256, 256, 0, stream>>>(W, wb, n8);
    }

    // 2) WMMA GEMM + fused activations: 512 m-tiles x 3 n-groups of 8 waves
    qrnn_gemm_gates<<<512 * 3, 256, 0, stream>>>(xb, wb, bias, yb);

    // 3) segmented scan: pass1 summaries, combine, pass2 emit
    qrnn_seg_scan<<<(NSEG * NCHAIN) / 256, 256, 0, stream>>>(yb, p_end, P_end);
    qrnn_seg_combine<<<NCHAIN / 256, 256, 0, stream>>>(p_end, P_end, seg_init);
    qrnn_seg_emit<<<(NSEG * NCHAIN) / 256, 256, 0, stream>>>(yb, seg_init, h_out, c_last);
}
